// RNNLayer_6640019440021
// MI455X (gfx1250) — matrix operand order repaired and checked by reading, not compiled, not run on hardware
//
#include <hip/hip_runtime.h>

// ---- ext-vector types (register-resident; avoid struct float4 -> scratch) ----
typedef __attribute__((ext_vector_type(16))) __bf16 v16bf;
typedef __attribute__((ext_vector_type(8)))  __bf16 v8bf;
typedef __attribute__((ext_vector_type(8)))  float  v8f;
typedef __attribute__((ext_vector_type(4)))  float  fvec4;

#define T_STEPS 12
#define NSEQ    56512      // B*N = 64*883
#define HSTRIDE 72         // bf16 elements per h-row in LDS (16B-aligned frags, conflict-free)
#define L2E     1.4426950408889634f
#define L2E2    2.8853900817779268f

// Cap VGPRs so >=4 waves/SIMD fit: working set ~170 VGPRs (weights stay in LDS).
__global__ __launch_bounds__(128, 4) void gru_fused(
    const float* __restrict__ X,     // [B=64, T=12, N=883, H=64]
    const float* __restrict__ Wih,   // [192, 64]
    const float* __restrict__ Whh,   // [192, 64]
    const float* __restrict__ bih,   // [192]
    const float* __restrict__ bhh,   // [192]
    float* __restrict__ out)         // [T=12, B*N=56512, H=64]
{
  // B-matrix fragments for both weight matrices, bf16, WMMA-ready:
  // index = ((mat*12 + ntile)*2 + kb)*32 + lane ; each lane holds K = kb*32 + (lane/16)*8 + {0..7, 16..23}.
  __shared__ v16bf wfrag[2 * 12 * 2 * 32];                 // 48 KB
  __shared__ alignas(16) __bf16 hb[4][16 * HSTRIDE];       // 9 KB: per-wave h scratch

  const int tid  = threadIdx.x;
  const int lane = tid & 31;
  const int wv   = tid >> 5;      // wave in workgroup (0..3)
  const int l16  = lane & 15;
  const int half = lane >> 4;     // 0: lanes 0-15, 1: lanes 16-31

  // ---- stage weights into LDS as bf16 B-fragments (once per workgroup) ----
  #pragma unroll
  for (int it = 0; it < 12; ++it) {
    int cid   = tid + it * 128;           // 0..1535
    int flane = cid & 31;
    int fkb   = (cid >> 5) & 1;
    int fnt   = (cid >> 6) % 12;
    int fmat  = cid / 768;
    const float* Wsrc = fmat ? Whh : Wih;
    // B[k][n] = W[ntile*16+n][k]; lane holds K = kb*32 + (lane/16)*8 + {0..7, 16..23} of column n=lane%16
    const float* p = Wsrc + (fnt * 16 + (flane & 15)) * 64 + fkb * 32 + (flane >> 4) * 8;
    v16bf v;
    #pragma unroll
    for (int j = 0; j < 16; ++j) v[j] = (__bf16)p[j < 8 ? j : j + 8];
    wfrag[cid] = v;
  }
  // zero h scratch (h0 = 0)
  {
    unsigned int* hz = (unsigned int*)&hb[0][0];
    for (int i = tid; i < 4 * 16 * HSTRIDE / 2; i += 128) hz[i] = 0u;
  }
  __syncthreads();

  // ---- per-lane bias constants, pre-folded into activation scaling ----
  // r/z: sigmoid(acc+b) = rcp(1 + exp2(fma(acc, -log2e, -log2e*b)))
  // n:   tanh(xn + bx + r*(hn + bh)) ; -2log2e*bx pre-multiplied
  float nbR[4], nbZ[4], bhn[4], nbx2[4];
  #pragma unroll
  for (int c = 0; c < 4; ++c) {
    nbR[c]  = -L2E  * (bih[c * 16 + l16]        + bhh[c * 16 + l16]);
    nbZ[c]  = -L2E  * (bih[(c + 4) * 16 + l16]  + bhh[(c + 4) * 16 + l16]);
    bhn[c]  =          bhh[128 + c * 16 + l16];
    nbx2[c] = -L2E2 *  bih[128 + c * 16 + l16];
  }

  // ---- sequence tile: 16 rows per wave ----
  const int m0    = blockIdx.x * 64 + wv * 16;
  const int myrow = m0 + l16;            // this lane's sequence (A-matrix row)
  const int bb    = myrow / 883;
  const int nn    = myrow % 883;
  // X element index = ((bb*12 + t)*883 + nn)*64 + k ; per-t stride = 883*64 = NSEQ
  const float* xlane = X + ((size_t)bb * 12 * 883 + nn) * 64 + half * 8;

  // h in C-layout registers (f32): hreg[c][i] = h[seq = i + 8*half][col = c*16 + l16]
  v8f hreg[4];
  #pragma unroll
  for (int c = 0; c < 4; ++c)
    #pragma unroll
    for (int i = 0; i < 8; ++i) hreg[c][i] = 0.0f;

  // prefetch x row chunks for t=0 (per lane: K = half*8..+7, +16..+23, per kb)
  fvec4 xf[8];
  {
    const float* p = xlane;
    xf[0] = *(const fvec4*)(p +  0);  xf[1] = *(const fvec4*)(p +  4);
    xf[2] = *(const fvec4*)(p + 16);  xf[3] = *(const fvec4*)(p + 20);
    xf[4] = *(const fvec4*)(p + 32);  xf[5] = *(const fvec4*)(p + 36);
    xf[6] = *(const fvec4*)(p + 48);  xf[7] = *(const fvec4*)(p + 52);
  }

  // loop-carried pointers (strength-reduced; no per-step 64-bit mul chains)
  const float* xpref = xlane + NSEQ;                               // x row for t+1
  float*       ob    = out + ((size_t)m0 + half * 8) * 64 + l16;   // out row for t
  const __bf16* hld  = &hb[wv][l16 * HSTRIDE + half * 8];          // h A-frag base
  __bf16*       hst  = &hb[wv][half * 8 * HSTRIDE + l16];          // h store base

  // Opaque zero, re-defined every iteration: blocks LICM of the weight-fragment
  // LDS loads (otherwise the compiler promotes ~384 VGPRs of weights and spills).
  unsigned wdyn = 0;

  for (int t = 0; t < T_STEPS; ++t) {
    asm volatile("" : "+v"(wdyn));          // compiler must assume wdyn changed
    const v16bf* wf = wfrag + wdyn;

    // convert prefetched x -> bf16 A-fragments (elements 0-7 = chunk0, 8-15 = chunk1)
    v16bf ax[2];
    #pragma unroll
    for (int kb = 0; kb < 2; ++kb)
      #pragma unroll
      for (int j = 0; j < 16; ++j)
        ax[kb][j] = (__bf16)xf[kb * 4 + (j >> 2)][j & 3];

    // prefetch x for t+1
    if (t < T_STEPS - 1) {
      const float* p = xpref;
      xf[0] = *(const fvec4*)(p +  0);  xf[1] = *(const fvec4*)(p +  4);
      xf[2] = *(const fvec4*)(p + 16);  xf[3] = *(const fvec4*)(p + 20);
      xf[4] = *(const fvec4*)(p + 32);  xf[5] = *(const fvec4*)(p + 36);
      xf[6] = *(const fvec4*)(p + 48);  xf[7] = *(const fvec4*)(p + 52);
      xpref += NSEQ;
    }

    // load h A-fragments from LDS (bf16, stored last step)
    v16bf ah[2];
    #pragma unroll
    for (int kb = 0; kb < 2; ++kb) {
      const __bf16* hp = hld + kb * 32;
      v8bf c0 = *(const v8bf*)(hp);
      v8bf c1 = *(const v8bf*)(hp + 16);
      ah[kb] = __builtin_shufflevector(c0, c1, 0, 1, 2, 3, 4, 5, 6, 7,
                                                8, 9, 10, 11, 12, 13, 14, 15);
    }

    #pragma unroll
    for (int c = 0; c < 4; ++c) {
      // accumulators start at inline 0 (no bias splats, no C-operand registers)
      v8f accR = {}, accZ = {}, accNx = {}, accNh = {};
      // gates tile c: r = ntile c, z = ntile c+4, n = ntile c+8 ; K=64 (2 kb) per matrix
      #pragma unroll
      for (int kb = 0; kb < 2; ++kb) {
        v16bf bR0 = wf[((0 * 12 + c) * 2 + kb) * 32 + lane];
        accR  = __builtin_amdgcn_wmma_f32_16x16x32_bf16(false, ax[kb], false, bR0, (short)0, accR,  false, false);
        v16bf bR1 = wf[((1 * 12 + c) * 2 + kb) * 32 + lane];
        accR  = __builtin_amdgcn_wmma_f32_16x16x32_bf16(false, ah[kb], false, bR1, (short)0, accR,  false, false);
        v16bf bZ0 = wf[((0 * 12 + c + 4) * 2 + kb) * 32 + lane];
        accZ  = __builtin_amdgcn_wmma_f32_16x16x32_bf16(false, ax[kb], false, bZ0, (short)0, accZ,  false, false);
        v16bf bZ1 = wf[((1 * 12 + c + 4) * 2 + kb) * 32 + lane];
        accZ  = __builtin_amdgcn_wmma_f32_16x16x32_bf16(false, ah[kb], false, bZ1, (short)0, accZ,  false, false);
        v16bf bN0 = wf[((0 * 12 + c + 8) * 2 + kb) * 32 + lane];
        accNx = __builtin_amdgcn_wmma_f32_16x16x32_bf16(false, ax[kb], false, bN0, (short)0, accNx, false, false);
        v16bf bN1 = wf[((1 * 12 + c + 8) * 2 + kb) * 32 + lane];
        accNh = __builtin_amdgcn_wmma_f32_16x16x32_bf16(false, ah[kb], false, bN1, (short)0, accNh, false, false);
      }
      // elementwise GRU gates (f32), biases folded into FMA constants
      #pragma unroll
      for (int i = 0; i < 8; ++i) {
        float eR = __builtin_amdgcn_exp2f(__builtin_fmaf(accR[i], -L2E, nbR[c]));
        float r  = __builtin_amdgcn_rcpf(1.0f + eR);
        float eZ = __builtin_amdgcn_exp2f(__builtin_fmaf(accZ[i], -L2E, nbZ[c]));
        float z  = __builtin_amdgcn_rcpf(1.0f + eZ);
        float a  = accNh[i] + bhn[c];                       // hn + b_hh_n
        float v  = __builtin_fmaf(r, a, accNx[i]);          // xn' + r*hn'
        float eN = __builtin_amdgcn_exp2f(__builtin_fmaf(v, -L2E2, nbx2[c]));
        float n  = __builtin_fmaf(2.0f, __builtin_amdgcn_rcpf(1.0f + eN), -1.0f);
        float h  = n + z * (hreg[c][i] - n);                // (1-z)*n + z*h_old
        hreg[c][i] = h;
        hst[i * HSTRIDE + c * 16] = (__bf16)h;              // bf16 view for next GEMM
        ob[(size_t)i * 64 + c * 16] = h;                    // f32 output
      }
    }
    ob += (size_t)NSEQ * 64;                                // advance one time step
  }
}

extern "C" void kernel_launch(void* const* d_in, const int* in_sizes, int n_in,
                              void* d_out, int out_size, void* d_ws, size_t ws_size,
                              hipStream_t stream) {
  const float* X   = (const float*)d_in[0];
  const float* Wih = (const float*)d_in[1];
  const float* Whh = (const float*)d_in[2];
  const float* bih = (const float*)d_in[3];
  const float* bhh = (const float*)d_in[4];
  float* out = (float*)d_out;
  (void)in_sizes; (void)n_in; (void)out_size; (void)d_ws; (void)ws_size;
  gru_fused<<<dim3(NSEQ / 64), dim3(128), 0, stream>>>(X, Wih, Whh, bih, bhh, out);
}
